// SpikeFP16MatMul_84713934946542
// MI455X (gfx1250) — matrix operand order repaired and checked by reading, not compiled, not run on hardware
//
#include <hip/hip_runtime.h>

typedef __attribute__((ext_vector_type(16))) _Float16 v16h;
typedef __attribute__((ext_vector_type(8)))  float    v8f;
typedef __attribute__((ext_vector_type(4)))  float    v4f;

#define BH_TOTAL 32      // 4 batches * 8 heads
#define MM 256
#define NN 256
#define KK 64
#define TILE 128
#define LDSS 66          // padded row stride in halves (66*2B=132B -> conflict-free column gathers)

// Decode 16 MSB-first pulse floats (0.0f/1.0f) into an fp16 bit pattern.
// pulse float is 0x00000000 or 0x3F800000 -> bit = (u >> 23) & 1 (pure integer ALU).
__device__ __forceinline__ unsigned short decode16(const uint4* __restrict__ q) {
  unsigned v = 0;
#pragma unroll
  for (int i = 0; i < 4; ++i) {
    uint4 u = q[i];
    v = (v << 1) | ((u.x >> 23) & 1u);
    v = (v << 1) | ((u.y >> 23) & 1u);
    v = (v << 1) | ((u.z >> 23) & 1u);
    v = (v << 1) | ((u.w >> 23) & 1u);
  }
  return (unsigned short)v;
}

__global__ __launch_bounds__(256)
void SpikeFP16MatMul_gemm_bits(const float* __restrict__ Abits,
                               const float* __restrict__ Bbits,
                               float* __restrict__ Out) {
  __shared__ unsigned short Asm[TILE * LDSS];
  __shared__ unsigned short Bsm[TILE * LDSS];

  const int tid  = threadIdx.x;
  const int bh   = blockIdx.x >> 2;       // 0..31
  const int tile = blockIdx.x & 3;        // 2x2 tiles of 128x128
  const int tm   = (tile >> 1) * TILE;
  const int tn   = (tile & 1) * TILE;

  const float* __restrict__ Abase = Abits + (size_t)bh * MM * KK * 16;
  const float* __restrict__ Bbase = Bbits + (size_t)bh * NN * KK * 16;

  // ---- Stage 1: cooperative bit-plane decode of A/B tiles into LDS fp16 ----
  for (int e = tid; e < TILE * KK; e += 256) {
    const int m = e >> 6;       // 0..127 (row within tile)
    const int k = e & 63;       // 0..63
    Asm[m * LDSS + k] = decode16(
        reinterpret_cast<const uint4*>(Abase + ((size_t)(tm + m) * KK + k) * 16));
    Bsm[m * LDSS + k] = decode16(
        reinterpret_cast<const uint4*>(Bbase + ((size_t)(tn + m) * KK + k) * 16));
  }
  __syncthreads();

  // ---- Stage 2: WMMA GEMM. 8 waves, each owns one 16-row block of the tile ----
  const int wv   = tid >> 5;    // 0..7  (row block)
  const int lane = tid & 31;
  const int rl   = lane & 15;
  const int hi   = lane >> 4;   // 0/1

  // A fragments (16x32 fp16, §7.12.2): lane L row = L&15,
  // halves 0..7 -> K = kb+0..7, halves 8..15 -> K = kb+16..23, kb = 8*(L>>4).
  v16h a0, a1;
  {
    const int row = wv * 16 + rl;
    const int kb  = hi * 8;
#pragma unroll
    for (int h = 0; h < 16; ++h) {
      const int koff = kb + ((h < 8) ? h : (h + 8));
      union { unsigned short u; _Float16 f; } c0, c1;
      c0.u = Asm[row * LDSS + koff];        // K step 0: K in [0,32)
      c1.u = Asm[row * LDSS + 32 + koff];   // K step 1: K in [32,64)
      a0[h] = c0.f;
      a1[h] = c1.f;
    }
  }

#pragma unroll
  for (int cb = 0; cb < 8; ++cb) {
    // B fragment (32x16, K x N): lane L col = L&15, half h -> K = 8*(L>>4) + (h<8 ? h : h+8).
    // B is stored [N,K] row-major, and C = A@B^T, so B_mat[k][n] = B[n][k].
    v16h b0, b1;
    const int col = cb * 16 + rl;
    const int kb2 = hi * 8;
#pragma unroll
    for (int h = 0; h < 16; ++h) {
      union { unsigned short u; _Float16 f; } c0, c1;
      c0.u = Bsm[col * LDSS + kb2 + ((h < 8) ? h : (h + 8))];
      c1.u = Bsm[col * LDSS + 32 + kb2 + ((h < 8) ? h : (h + 8))];
      b0[h] = c0.f;
      b1[h] = c1.f;
    }

    v8f acc = {};
    acc = __builtin_amdgcn_wmma_f32_16x16x32_f16(false, a0, false, b0,
                                                 (short)0, acc, false, false);
    acc = __builtin_amdgcn_wmma_f32_16x16x32_f16(false, a1, false, b1,
                                                 (short)0, acc, false, false);

    // ---- Stage 3: fp32 -> 32 pulse bits, streamed with NT b128 stores ----
    // C/D layout (§7.12.2): lane L, VGPR j -> row = 8*(L>>4)+j, col = L&15.
    const int gn = tn + cb * 16 + rl;
#pragma unroll
    for (int j = 0; j < 8; ++j) {
      const int gm = tm + wv * 16 + hi * 8 + j;
      const unsigned bits = __float_as_uint(acc[j]);
      v4f* __restrict__ dst =
          reinterpret_cast<v4f*>(Out + (((size_t)bh * MM + gm) * NN + gn) * 32);
#pragma unroll
      for (int g = 0; g < 8; ++g) {
        v4f f;
        f.x = (float)((bits >> (31 - 4 * g)) & 1u);
        f.y = (float)((bits >> (30 - 4 * g)) & 1u);
        f.z = (float)((bits >> (29 - 4 * g)) & 1u);
        f.w = (float)((bits >> (28 - 4 * g)) & 1u);
        __builtin_nontemporal_store(f, &dst[g]);   // 268MB stream > L2: bypass
      }
    }
  }
}

extern "C" void kernel_launch(void* const* d_in, const int* in_sizes, int n_in,
                              void* d_out, int out_size, void* d_ws, size_t ws_size,
                              hipStream_t stream) {
  (void)in_sizes; (void)n_in; (void)d_ws; (void)ws_size; (void)out_size;
  const float* Abits = (const float*)d_in[0];  // [4,8,256,64,16] fp16 pulses
  const float* Bbits = (const float*)d_in[1];  // [4,8,256,64,16] fp16 pulses
  float* Out = (float*)d_out;                  // [4,8,256,256,32] fp32 pulses

  dim3 grid(BH_TOTAL * 4);   // 32 (b,h) x 4 (128x128 C tiles)
  dim3 block(256);           // 8 wave32s
  SpikeFP16MatMul_gemm_bits<<<grid, block, 0, stream>>>(Abits, Bbits, Out);
}
